// HSTUEncoder_24026047054164
// MI455X (gfx1250) — matrix operand order repaired and checked by reading, not compiled, not run on hardware
//
#include <hip/hip_runtime.h>
#include <hip/hip_bf16.h>

// ---------------- CDNA5 WMMA types ----------------
typedef __attribute__((ext_vector_type(16))) __bf16 v16bf;
typedef __attribute__((ext_vector_type(8)))  float  v8f;

#define D_MODEL 512
#define SEQ_N   1024
#define NHEAD   8
#define HDIM    64
#define NBATCH  4
#define NBLK    4
#define UVQK_N  2048

static __device__ __forceinline__ unsigned short f2bf(float x) {
  unsigned int u = __float_as_uint(x);
  unsigned int r = (u + 0x7FFFu + ((u >> 16) & 1u)) >> 16;   // RNE
  return (unsigned short)r;
}

static __device__ __forceinline__ float silu(float x) {
  return x / (1.0f + __expf(-x));
}

// A-fragment: 16x32 bf16, source row-major [row][k], lda in elements.
// lane: r=lane&15 (row), hi=lane>>4; elems 0..15 = K=hi*16+{0..15} contiguous
static __device__ __forceinline__ v16bf load_afrag(const unsigned short* base, int lda, int koff) {
  int lane = threadIdx.x & 31;
  int r = lane & 15, hi = lane >> 4;
  const unsigned short* p = base + (size_t)r * lda + koff + hi * 16;
  union { v16bf v; uint4 u[2]; } f;
  f.u[0] = *reinterpret_cast<const uint4*>(p);
  f.u[1] = *reinterpret_cast<const uint4*>(p + 8);
  return f.v;
}

// B-fragment: 32x16 bf16, source n-major / K-contiguous: base points at column-row n0,
// ldn = stride between n rows (elements). lane: n=lane&15, K = hi*16 + {0..15} contiguous.
static __device__ __forceinline__ v16bf load_bfrag(const unsigned short* base, int ldn, int koff) {
  int lane = threadIdx.x & 31;
  int n = lane & 15, hi = lane >> 4;
  const unsigned short* p = base + (size_t)n * ldn + koff + hi * 16;
  union { v16bf v; uint4 u[2]; } f;
  f.u[0] = *reinterpret_cast<const uint4*>(p);
  f.u[1] = *reinterpret_cast<const uint4*>(p + 8);
  return f.v;
}

static __device__ __forceinline__ v8f wmma_bf16(v16bf a, v16bf b, v8f c) {
  return __builtin_amdgcn_wmma_f32_16x16x32_bf16(false, a, false, b, (short)0, c, false, false);
}

// ---------------- elementwise / LN kernels ----------------

__device__ __forceinline__ float block_reduce_sum256(float v, float* red) {
  int tid = threadIdx.x;
  red[tid] = v; __syncthreads();
  for (int s = 128; s > 0; s >>= 1) {
    if (tid < s) red[tid] += red[tid + s];
    __syncthreads();
  }
  float r = red[0];
  __syncthreads();
  return r;
}

__global__ void __launch_bounds__(256) embed_kernel(const float* __restrict__ seq,
                                                    const int* __restrict__ slen,
                                                    const float* __restrict__ pe,
                                                    float* __restrict__ x) {
  const int total = NBATCH * SEQ_N * D_MODEL;
  for (int idx = blockIdx.x * blockDim.x + threadIdx.x; idx < total;
       idx += gridDim.x * blockDim.x) {
    int d = idx & (D_MODEL - 1);
    int n = (idx >> 9) & (SEQ_N - 1);
    int b = idx >> 19;
    float m = (n < slen[b]) ? 1.0f : 0.0f;
    x[idx] = (seq[idx] * 22.627416997969522f + pe[n * D_MODEL + d]) * m;
  }
}

// transpose-convert fp32 [L][K][Nc] -> bf16 [L][Nc][K]
__global__ void __launch_bounds__(256) convT_kernel(const float* __restrict__ in,
                                                    unsigned short* __restrict__ out,
                                                    int K, int Nc, int total) {
  int per = K * Nc;
  for (int idx = blockIdx.x * blockDim.x + threadIdx.x; idx < total;
       idx += gridDim.x * blockDim.x) {
    int l = idx / per;
    int rm = idx - l * per;
    int kk = rm / Nc;
    int n = rm - kk * Nc;
    out[(size_t)l * per + (size_t)n * K + kk] = f2bf(in[idx]);
  }
}

__global__ void __launch_bounds__(256) ln1_kernel(const float* __restrict__ x,
                                                  const float* __restrict__ g,
                                                  const float* __restrict__ b,
                                                  unsigned short* __restrict__ nx) {
  __shared__ float red[256];
  int t = blockIdx.x, tid = threadIdx.x;
  const float* row = x + (size_t)t * D_MODEL;
  float v0 = row[tid], v1 = row[tid + 256];
  float mu = block_reduce_sum256(v0 + v1, red) * (1.0f / 512.0f);
  float d0 = v0 - mu, d1 = v1 - mu;
  float var = block_reduce_sum256(d0 * d0 + d1 * d1, red) * (1.0f / 512.0f);
  float rstd = rsqrtf(var + 1e-6f);
  nx[(size_t)t * D_MODEL + tid]       = f2bf(d0 * rstd * g[tid] + b[tid]);
  nx[(size_t)t * D_MODEL + tid + 256] = f2bf(d1 * rstd * g[tid + 256] + b[tid + 256]);
}

__global__ void __launch_bounds__(256) ln2u_kernel(const float* __restrict__ av,
                                                   const float* __restrict__ g,
                                                   const float* __restrict__ b,
                                                   const float* __restrict__ u,
                                                   unsigned short* __restrict__ oin) {
  __shared__ float red[256];
  int t = blockIdx.x, tid = threadIdx.x;
  const float* row = av + (size_t)t * D_MODEL;
  float v0 = row[tid], v1 = row[tid + 256];
  float mu = block_reduce_sum256(v0 + v1, red) * (1.0f / 512.0f);
  float d0 = v0 - mu, d1 = v1 - mu;
  float var = block_reduce_sum256(d0 * d0 + d1 * d1, red) * (1.0f / 512.0f);
  float rstd = rsqrtf(var + 1e-6f);
  const float* ur = u + (size_t)t * D_MODEL;
  oin[(size_t)t * D_MODEL + tid]       = f2bf((d0 * rstd * g[tid] + b[tid]) * ur[tid]);
  oin[(size_t)t * D_MODEL + tid + 256] = f2bf((d1 * rstd * g[tid + 256] + b[tid + 256]) * ur[tid + 256]);
}

__global__ void __launch_bounds__(256) l2norm_kernel(const float* __restrict__ x,
                                                     float* __restrict__ out) {
  __shared__ float red[256];
  int t = blockIdx.x, tid = threadIdx.x;
  const float* row = x + (size_t)t * D_MODEL;
  float v0 = row[tid], v1 = row[tid + 256];
  float ss = block_reduce_sum256(v0 * v0 + v1 * v1, red);
  float scale = 1.0f / fmaxf(sqrtf(ss), 1e-6f);
  out[(size_t)t * D_MODEL + tid]       = v0 * scale;
  out[(size_t)t * D_MODEL + tid + 256] = v1 * scale;
}

// ---------------- uvqk GEMM: silu(nx[4096x512] @ W[512x2048]) ----------------
// 4 waves/block, wave tile 16x64, software-pipelined K loop (double-buffered
// fragments so next-iteration loads are in flight during the WMMA group).
__global__ void __launch_bounds__(128) gemm_uvqk_kernel(const unsigned short* __restrict__ nx,
                                                        const unsigned short* __restrict__ wT,
                                                        float* __restrict__ u_out,
                                                        unsigned short* __restrict__ vT_out,
                                                        unsigned short* __restrict__ q_out,
                                                        unsigned short* __restrict__ k_out) {
  int w = threadIdx.x >> 5;
  int lane = threadIdx.x & 31;
  int hi = lane >> 4, lc = lane & 15;
  int m0 = (blockIdx.y * 4 + w) * 16;
  int n0 = blockIdx.x * 64;
  v8f acc0 = {}, acc1 = {}, acc2 = {}, acc3 = {};
  const unsigned short* abase = nx + (size_t)m0 * D_MODEL;
  const unsigned short* b0 = wT + (size_t)(n0 +  0) * D_MODEL;
  const unsigned short* b1 = wT + (size_t)(n0 + 16) * D_MODEL;
  const unsigned short* b2 = wT + (size_t)(n0 + 32) * D_MODEL;
  const unsigned short* b3 = wT + (size_t)(n0 + 48) * D_MODEL;

  v16bf a  = load_afrag(abase, D_MODEL, 0);
  v16bf f0 = load_bfrag(b0, D_MODEL, 0);
  v16bf f1 = load_bfrag(b1, D_MODEL, 0);
  v16bf f2 = load_bfrag(b2, D_MODEL, 0);
  v16bf f3 = load_bfrag(b3, D_MODEL, 0);
  for (int kk = 0; kk < D_MODEL - 32; kk += 32) {
    int kn = kk + 32;
    __builtin_prefetch(b0 + (size_t)lc * D_MODEL + kn + 32, 0, 1);  // global_prefetch_b8
    v16bf an = load_afrag(abase, D_MODEL, kn);
    v16bf g0 = load_bfrag(b0, D_MODEL, kn);
    v16bf g1 = load_bfrag(b1, D_MODEL, kn);
    v16bf g2 = load_bfrag(b2, D_MODEL, kn);
    v16bf g3 = load_bfrag(b3, D_MODEL, kn);
    acc0 = wmma_bf16(a, f0, acc0);
    acc1 = wmma_bf16(a, f1, acc1);
    acc2 = wmma_bf16(a, f2, acc2);
    acc3 = wmma_bf16(a, f3, acc3);
    a = an; f0 = g0; f1 = g1; f2 = g2; f3 = g3;
  }
  acc0 = wmma_bf16(a, f0, acc0);
  acc1 = wmma_bf16(a, f1, acc1);
  acc2 = wmma_bf16(a, f2, acc2);
  acc3 = wmma_bf16(a, f3, acc3);

  int seg = n0 >> 9;  // uniform for the whole block
  v8f accs[4] = {acc0, acc1, acc2, acc3};
  #pragma unroll
  for (int t = 0; t < 4; ++t) {
    int ncol = n0 + t * 16 + lc;
    int c512 = ncol & 511;
    #pragma unroll
    for (int g = 0; g < 8; ++g) {
      int m = m0 + g + 8 * hi;
      float sv = silu(accs[t][g]);
      if (seg == 0) {
        u_out[(size_t)m * D_MODEL + c512] = sv;
      } else if (seg == 1) {
        int hh = c512 >> 6, dv = c512 & 63;
        int bb = m >> 10, ns = m & 1023;
        vT_out[(((size_t)bb * NHEAD + hh) * HDIM + dv) * SEQ_N + ns] = f2bf(sv);
      } else if (seg == 2) {
        q_out[(size_t)m * D_MODEL + c512] = f2bf(sv);
      } else {
        k_out[(size_t)m * D_MODEL + c512] = f2bf(sv);
      }
    }
  }
}

// ---------------- fused HSTU attention ----------------
// One wave per (b, h, 16-row query tile). Flash-style over 32-wide key pairs.
// S = q@k^T (bf16 WMMA, K=64) ; P = silu(S + relbias) * causal/N ; av += P @ v.
// S(C-frag) -> P(A-frag) via 16x32 per-wave LDS scratch (no barrier needed).
// v-tile B-fragments are loaded right after the S WMMAs so they overlap the
// SiLU/bias elementwise work and the LDS round-trip.
__global__ void __launch_bounds__(128) attn_kernel(const unsigned short* __restrict__ qb,
                                                   const unsigned short* __restrict__ kb,
                                                   const unsigned short* __restrict__ vT,
                                                   const float* __restrict__ pwl,
                                                   float* __restrict__ av) {
  __shared__ __align__(16) unsigned short plds[4][16 * 32];
  int w = threadIdx.x >> 5;
  int lane = threadIdx.x & 31;
  int hi = lane >> 4, lc = lane & 15;
  int wg = blockIdx.x * 4 + w;
  int it = wg & 63;
  int h = (wg >> 6) & 7;
  int b = wg >> 9;
  int i0 = it * 16;

  const unsigned short* qbase = qb + ((size_t)(b * SEQ_N + i0)) * D_MODEL + h * HDIM;
  v16bf a0 = load_afrag(qbase, D_MODEL, 0);
  v16bf a1 = load_afrag(qbase, D_MODEL, 32);

  v8f o0 = {}, o1 = {}, o2 = {}, o3 = {};
  unsigned short* myp = plds[w];
  const unsigned short* vbase = vT + (((size_t)b * NHEAD + h) * HDIM) * SEQ_N;
  const float invN = 1.0f / (float)SEQ_N;

  for (int j0 = 0; j0 <= i0 + 15; j0 += 32) {
    const unsigned short* kb0 = kb + ((size_t)(b * SEQ_N + j0)) * D_MODEL + h * HDIM;
    const unsigned short* kb1 = kb0 + 16 * D_MODEL;
    v8f s0 = {}, s1 = {};
    s0 = wmma_bf16(a0, load_bfrag(kb0, D_MODEL, 0),  s0);
    s0 = wmma_bf16(a1, load_bfrag(kb0, D_MODEL, 32), s0);
    s1 = wmma_bf16(a0, load_bfrag(kb1, D_MODEL, 0),  s1);
    s1 = wmma_bf16(a1, load_bfrag(kb1, D_MODEL, 32), s1);
    // issue v-tile loads now; they are independent of the elementwise section
    v16bf vb0 = load_bfrag(vbase +  0 * SEQ_N, SEQ_N, j0);
    v16bf vb1 = load_bfrag(vbase + 16 * SEQ_N, SEQ_N, j0);
    v16bf vb2 = load_bfrag(vbase + 32 * SEQ_N, SEQ_N, j0);
    v16bf vb3 = load_bfrag(vbase + 48 * SEQ_N, SEQ_N, j0);
    #pragma unroll
    for (int g = 0; g < 8; ++g) {
      int ig = i0 + g + 8 * hi;                 // query index
      int j = j0 + lc;                          // key index, tile 0
      float p0 = s0[g] + pwl[j - ig + (SEQ_N - 1)];
      p0 = silu(p0) * ((j <= ig) ? invN : 0.0f);
      myp[(g + 8 * hi) * 32 + lc] = f2bf(p0);
      j = j0 + 16 + lc;                         // tile 1
      float p1 = s1[g] + pwl[j - ig + (SEQ_N - 1)];
      p1 = silu(p1) * ((j <= ig) ? invN : 0.0f);
      myp[(g + 8 * hi) * 32 + 16 + lc] = f2bf(p1);
    }
    // wave-private LDS round-trip: C-frag layout -> A-frag layout
    v16bf p = load_afrag(myp, 32, 0);
    o0 = wmma_bf16(p, vb0, o0);
    o1 = wmma_bf16(p, vb1, o1);
    o2 = wmma_bf16(p, vb2, o2);
    o3 = wmma_bf16(p, vb3, o3);
  }

  float* avb = av + ((size_t)(b * SEQ_N + i0)) * D_MODEL + h * HDIM;
  #pragma unroll
  for (int g = 0; g < 8; ++g) {
    size_t rr = (size_t)(g + 8 * hi) * D_MODEL;
    avb[rr +  0 + lc] = o0[g];
    avb[rr + 16 + lc] = o1[g];
    avb[rr + 32 + lc] = o2[g];
    avb[rr + 48 + lc] = o3[g];
  }
}

// ---------------- output GEMM + residual + mask ----------------
// x[4096x512] = (x + oin[4096x512] @ W_o[512x512] + b_o) * mask
__global__ void __launch_bounds__(128) gemm_wo_kernel(const unsigned short* __restrict__ oin,
                                                      const unsigned short* __restrict__ wT,
                                                      const float* __restrict__ b_o,
                                                      const int* __restrict__ slen,
                                                      float* __restrict__ x) {
  int w = threadIdx.x >> 5;
  int lane = threadIdx.x & 31;
  int hi = lane >> 4, lc = lane & 15;
  int m0 = (blockIdx.y * 4 + w) * 16;
  int n0 = blockIdx.x * 64;
  v8f acc0 = {}, acc1 = {}, acc2 = {}, acc3 = {};
  const unsigned short* abase = oin + (size_t)m0 * D_MODEL;
  const unsigned short* b0 = wT + (size_t)(n0 +  0) * D_MODEL;
  const unsigned short* b1 = wT + (size_t)(n0 + 16) * D_MODEL;
  const unsigned short* b2 = wT + (size_t)(n0 + 32) * D_MODEL;
  const unsigned short* b3 = wT + (size_t)(n0 + 48) * D_MODEL;

  v16bf a  = load_afrag(abase, D_MODEL, 0);
  v16bf f0 = load_bfrag(b0, D_MODEL, 0);
  v16bf f1 = load_bfrag(b1, D_MODEL, 0);
  v16bf f2 = load_bfrag(b2, D_MODEL, 0);
  v16bf f3 = load_bfrag(b3, D_MODEL, 0);
  for (int kk = 0; kk < D_MODEL - 32; kk += 32) {
    int kn = kk + 32;
    v16bf an = load_afrag(abase, D_MODEL, kn);
    v16bf g0 = load_bfrag(b0, D_MODEL, kn);
    v16bf g1 = load_bfrag(b1, D_MODEL, kn);
    v16bf g2 = load_bfrag(b2, D_MODEL, kn);
    v16bf g3 = load_bfrag(b3, D_MODEL, kn);
    acc0 = wmma_bf16(a, f0, acc0);
    acc1 = wmma_bf16(a, f1, acc1);
    acc2 = wmma_bf16(a, f2, acc2);
    acc3 = wmma_bf16(a, f3, acc3);
    a = an; f0 = g0; f1 = g1; f2 = g2; f3 = g3;
  }
  acc0 = wmma_bf16(a, f0, acc0);
  acc1 = wmma_bf16(a, f1, acc1);
  acc2 = wmma_bf16(a, f2, acc2);
  acc3 = wmma_bf16(a, f3, acc3);

  v8f accs[4] = {acc0, acc1, acc2, acc3};
  #pragma unroll
  for (int t = 0; t < 4; ++t) {
    int ncol = n0 + t * 16 + lc;
    #pragma unroll
    for (int g = 0; g < 8; ++g) {
      int m = m0 + g + 8 * hi;
      float mval = ((m & 1023) < slen[m >> 10]) ? 1.0f : 0.0f;
      size_t xi = (size_t)m * D_MODEL + ncol;
      x[xi] = (x[xi] + accs[t][g] + b_o[ncol]) * mval;
    }
  }
}

// ---------------- host launch ----------------

#define X_OFF    ((size_t)0)           //  8 MiB fp32 x
#define NX_OFF   ((size_t)8388608)     //  4 MiB bf16 nx
#define U_OFF    ((size_t)12582912)    //  8 MiB fp32 u
#define Q_OFF    ((size_t)20971520)    //  4 MiB bf16 q
#define K_OFF    ((size_t)25165824)    //  4 MiB bf16 k
#define VT_OFF   ((size_t)29360128)    //  4 MiB bf16 vT
#define AV_OFF   ((size_t)33554432)    //  8 MiB fp32 av
#define OIN_OFF  ((size_t)41943040)    //  4 MiB bf16 o_in
#define WUV_OFF  ((size_t)46137344)    //  8 MiB bf16 w_uvqk^T
#define WO_OFF   ((size_t)54525952)    //  2 MiB bf16 w_o^T

extern "C" void kernel_launch(void* const* d_in, const int* in_sizes, int n_in,
                              void* d_out, int out_size, void* d_ws, size_t ws_size,
                              hipStream_t stream) {
  (void)in_sizes; (void)n_in; (void)out_size; (void)ws_size;
  const float* seq   = (const float*)d_in[0];
  const int*   slen  = (const int*)d_in[1];
  const float* pe    = (const float*)d_in[2];
  const float* ln1g  = (const float*)d_in[3];
  const float* ln1b  = (const float*)d_in[4];
  const float* wuvqk = (const float*)d_in[5];
  const float* ln2g  = (const float*)d_in[6];
  const float* ln2b  = (const float*)d_in[7];
  const float* wo    = (const float*)d_in[8];
  const float* bo    = (const float*)d_in[9];
  const float* posw  = (const float*)d_in[10];

  char* ws = (char*)d_ws;
  float*          x    = (float*)(ws + X_OFF);
  unsigned short* nx   = (unsigned short*)(ws + NX_OFF);
  float*          u    = (float*)(ws + U_OFF);
  unsigned short* qb   = (unsigned short*)(ws + Q_OFF);
  unsigned short* kbuf = (unsigned short*)(ws + K_OFF);
  unsigned short* vt   = (unsigned short*)(ws + VT_OFF);
  float*          av   = (float*)(ws + AV_OFF);
  unsigned short* oin  = (unsigned short*)(ws + OIN_OFF);
  unsigned short* wuvT = (unsigned short*)(ws + WUV_OFF);
  unsigned short* woT  = (unsigned short*)(ws + WO_OFF);

  // one-time (per call) weight transpose/convert to bf16
  convT_kernel<<<4096, 256, 0, stream>>>(wuvqk, wuvT, D_MODEL, UVQK_N, NBLK * D_MODEL * UVQK_N);
  convT_kernel<<<1024, 256, 0, stream>>>(wo, woT, D_MODEL, D_MODEL, NBLK * D_MODEL * D_MODEL);
  embed_kernel<<<2048, 256, 0, stream>>>(seq, slen, pe, x);

  for (int l = 0; l < NBLK; ++l) {
    ln1_kernel<<<NBATCH * SEQ_N, 256, 0, stream>>>(x, ln1g + l * D_MODEL, ln1b + l * D_MODEL, nx);
    gemm_uvqk_kernel<<<dim3(UVQK_N / 64, NBATCH * SEQ_N / 64), 128, 0, stream>>>(
        nx, wuvT + (size_t)l * D_MODEL * UVQK_N, u, vt, qb, kbuf);
    attn_kernel<<<NBATCH * NHEAD * (SEQ_N / 16) / 4, 128, 0, stream>>>(
        qb, kbuf, vt, posw + l * (2 * SEQ_N - 1), av);
    ln2u_kernel<<<NBATCH * SEQ_N, 256, 0, stream>>>(av, ln2g + l * D_MODEL, ln2b + l * D_MODEL, u, oin);
    gemm_wo_kernel<<<dim3(D_MODEL / 64, NBATCH * SEQ_N / 64), 128, 0, stream>>>(
        oin, woT + (size_t)l * D_MODEL * D_MODEL, bo + l * D_MODEL, slen, x);
  }

  l2norm_kernel<<<NBATCH * SEQ_N, 256, 0, stream>>>(x, (float*)d_out);
}
